// HawkesKT_29343216566953
// MI455X (gfx1250) — matrix operand order repaired and checked by reading, not compiled, not run on hardware
//
#include <hip/hip_runtime.h>
#include <hip/hip_bf16.h>

typedef __attribute__((ext_vector_type(16))) _Float16 v16h;
typedef __attribute__((ext_vector_type(8)))  _Float16 v8h;
typedef __attribute__((ext_vector_type(8)))  float    v8f;

#define NSKILLS 1000
#define EMB     128
#define BB      64
#define LL      1024
// 1/ln(5): exp(-beta * ln(dt)/ln5) == exp2(-beta * log2(dt) / ln5)
#define INV_LN5 0.6213349345596119f
#define LOG2E   1.4426950408889634f

__global__ void cvt_f32_to_f16(const float* __restrict__ src,
                               _Float16* __restrict__ dst, int n) {
    int i = blockIdx.x * blockDim.x + threadIdx.x;
    if (i < n) dst[i] = (_Float16)src[i];
}

// Load the alpha/beta A-matrix fragments (16x32 f16 tile, 4 k-steps) for one i-tile.
// ISA A layout (16x16x32 f16): lanes 0-15: halves 0..7 = K 0..7, halves 8..15 = K 16..23;
// lanes 16-31: +8 on both K ranges (ka = 8*half).
#define LOAD_AFRAGS(Aa, Ab, roff)                                        \
    {                                                                    \
        const _Float16* rAa = aih + (roff);                              \
        const _Float16* rAb = bih + (roff);                              \
        _Pragma("unroll")                                                \
        for (int ks = 0; ks < 4; ++ks) {                                 \
            ((v8h*)&Aa[ks])[0] = *(const v8h*)(rAa + ks * 32 + ka);      \
            ((v8h*)&Aa[ks])[1] = *(const v8h*)(rAa + ks * 32 + ka + 16); \
            ((v8h*)&Ab[ks])[0] = *(const v8h*)(rAb + ks * 32 + ka);      \
            ((v8h*)&Ab[ks])[1] = *(const v8h*)(rAb + ks * 32 + ka + 16); \
        }                                                                \
    }

// 8 WMMAs (alpha + beta over K=128) then the masked elementwise Hawkes term.
// C/D layout: element g -> M = g + 8*half, N = lane&15.
#define COMPUTE_TILE(Aa, Ab, itile)                                           \
    {                                                                         \
        v8f ca = {}, cb = {};                                                 \
        _Pragma("unroll")                                                     \
        for (int ks = 0; ks < 4; ++ks) {                                      \
            ca = __builtin_amdgcn_wmma_f32_16x16x32_f16(false, Aa[ks], false, \
                     Ba[ks], (short)0, ca, false, false);                     \
            cb = __builtin_amdgcn_wmma_f32_16x16x32_f16(false, Ab[ks], false, \
                     Bb[ks], (short)0, cb, false, false);                     \
        }                                                                     \
        const int mbase = (itile) * 16 + half * 8;                            \
        _Pragma("unroll")                                                     \
        for (int g = 0; g < 8; ++g) {                                         \
            int   i  = mbase + g;                                             \
            float ti = t_lds[i];                                              \
            float bt = fminf(fmaxf(cb[g] + 1.0f, 0.0f), 10.0f);               \
            float dt = fabsf(ti - tj);                                        \
            float l2 = __builtin_amdgcn_logf(dt + 1e-10f);                    \
            float cr = ca[g] * __builtin_amdgcn_exp2f(-bt * l2 * INV_LN5);    \
            acc += (i < j) ? cr : 0.0f;                                       \
        }                                                                     \
    }

__launch_bounds__(128)
__global__ void hawkes_main(const int*   __restrict__ inp,          // (B,4,L) int32
                            const float* __restrict__ problem_base, // (20000,1)
                            const float* __restrict__ skill_base,   // (1000,1)
                            const _Float16* __restrict__ aih,       // alpha_inter f16 (2000,128)
                            const _Float16* __restrict__ ash,       // alpha_skill f16 (1000,128)
                            const _Float16* __restrict__ bih,       // beta_inter  f16 (2000,128)
                            const _Float16* __restrict__ bsh,       // beta_skill  f16 (1000,128)
                            float* __restrict__ out) {              // (B,L)
    __shared__ float t_lds[LL];      // times as f32 (exact: < 2^24)
    __shared__ int   irow_lds[LL];   // (skill + label*NSKILLS) * EMB

    const int b = blockIdx.y;
    const int* base = inp + b * 4 * LL;

    // Cooperative preload of this batch's sequence metadata
    for (int idx = threadIdx.x; idx < LL; idx += 128) {
        int s   = base[idx];
        int lab = base[2 * LL + idx];
        int t   = base[3 * LL + idx];
        irow_lds[idx] = (s + lab * NSKILLS) * EMB;
        t_lds[idx]    = (float)t;
    }
    __syncthreads();

    const int lane  = threadIdx.x & 31;
    const int wave  = threadIdx.x >> 5;
    const int jtile = blockIdx.x * 4 + wave;   // 0..63 (wave-uniform)
    const int n     = lane & 15;
    const int half  = lane >> 4;               // 0: lanes 0-15, 1: lanes 16-31
    const int j     = jtile * 16 + n;

    // ---- B fragments: skill embeddings of the 16 j-columns (fixed per wave) ----
    // ISA B layout (16x16x32 f16): lanes 0-15 hold K=ks*32+0..7,16..23 for N=lane,
    // lanes 16-31 hold K=ks*32+8..15,24..31 for N=lane-16.
    const int skill_j = base[j];
    const _Float16* rowBa = ash + skill_j * EMB;
    const _Float16* rowBb = bsh + skill_j * EMB;
    const int kb = half * 8;
    v16h Ba[4], Bb[4];
#pragma unroll
    for (int ks = 0; ks < 4; ++ks) {
        ((v8h*)&Ba[ks])[0] = *(const v8h*)(rowBa + ks * 32 + kb);
        ((v8h*)&Ba[ks])[1] = *(const v8h*)(rowBa + ks * 32 + kb + 16);
        ((v8h*)&Bb[ks])[0] = *(const v8h*)(rowBb + ks * 32 + kb);
        ((v8h*)&Bb[ks])[1] = *(const v8h*)(rowBb + ks * 32 + kb + 16);
    }

    const float tj = t_lds[j];
    const int   ka = half * 8;
    float acc = 0.0f;

    // ---- double-buffered i-tile loop: prefetch tile k+1's A fragments, then
    // run tile k's 8 WMMAs + transcendental elementwise while loads fly ----
    v16h Aa0[4], Ab0[4], Aa1[4], Ab1[4];
    {
        int r0 = irow_lds[n];                  // itile 0
        LOAD_AFRAGS(Aa0, Ab0, r0)
    }
    int itile = 0;
    for (;;) {
        {
            int nt = (itile + 1 <= jtile) ? (itile + 1) : jtile;   // clamp: tail reload
            int rn = irow_lds[nt * 16 + n];
            LOAD_AFRAGS(Aa1, Ab1, rn)
        }
        COMPUTE_TILE(Aa0, Ab0, itile)
        ++itile;
        if (itile > jtile) break;
        {
            int nt = (itile + 1 <= jtile) ? (itile + 1) : jtile;
            int rn = irow_lds[nt * 16 + n];
            LOAD_AFRAGS(Aa0, Ab0, rn)
        }
        COMPUTE_TILE(Aa1, Ab1, itile)
        ++itile;
        if (itile > jtile) break;
    }

    // lanes n and n+16 hold the same column j: combine halves
    acc += __shfl_xor(acc, 16, 32);

    if (lane < 16) {
        int   prob = base[LL + j];
        float x    = problem_base[prob] + skill_base[skill_j] + acc;
        float e    = __builtin_amdgcn_exp2f(-x * LOG2E);            // exp(-x)
        out[b * LL + j] = 1.0f / (1.0f + e);
    }
}

extern "C" void kernel_launch(void* const* d_in, const int* in_sizes, int n_in,
                              void* d_out, int out_size, void* d_ws, size_t ws_size,
                              hipStream_t stream) {
    const int*   inp = (const int*)  d_in[0];
    const float* pb  = (const float*)d_in[1];
    const float* sb  = (const float*)d_in[2];
    const float* ai  = (const float*)d_in[3];
    const float* as  = (const float*)d_in[4];
    const float* bi  = (const float*)d_in[5];
    const float* bs  = (const float*)d_in[6];
    float* out = (float*)d_out;

    // f16 staging tables in workspace (total 768000 halfs = 1.5 MB)
    _Float16* ws  = (_Float16*)d_ws;
    _Float16* aih = ws;                 // 2000*128
    _Float16* ash = ws + 256000;        // 1000*128
    _Float16* bih = ws + 384000;        // 2000*128
    _Float16* bsh = ws + 640000;        // 1000*128

    const int nai = 2000 * EMB;
    const int nas = 1000 * EMB;
    cvt_f32_to_f16<<<(nai + 255) / 256, 256, 0, stream>>>(ai, aih, nai);
    cvt_f32_to_f16<<<(nas + 255) / 256, 256, 0, stream>>>(as, ash, nas);
    cvt_f32_to_f16<<<(nai + 255) / 256, 256, 0, stream>>>(bi, bih, nai);
    cvt_f32_to_f16<<<(nas + 255) / 256, 256, 0, stream>>>(bs, bsh, nas);

    // 4 waves per block, each wave owns one 16-column j-tile; 16 groups x 64 batches
    dim3 grid(16, BB);
    hawkes_main<<<grid, 128, 0, stream>>>(inp, pb, sb, aih, ash, bih, bsh, out);
}
